// LSTMHierarchialAttention_61684320305478
// MI455X (gfx1250) — matrix operand order repaired and checked by reading, not compiled, not run on hardware
//
#include <hip/hip_runtime.h>
#include <hip/hip_bf16.h>

// ---------------------------------------------------------------------------
// LSTM with hierarchical attention, MI455X (gfx1250, wave32, WMMA).
// S=32, W=128, I=H=2048, 4H=8192, T=S*W=4096 sequential steps.
//
//  K1: fp32 -> bf16 conversions (inputs, W_ih, W_hh)
//  K2: softmax(attW, axis=1) transposed -> attn (W,H)
//  K3: Gx[t,n] = sum_k x[t,k]*W_ih[n,k] + b_ih[n] + b_hh[n]   (WMMA bf16 GEMM)
//  K4: persistent sequential scan: gates = Gx[t] + W_hh@h via WMMA matvec
//      (h in row 0 of A tile), cell update in fp32, grid-wide atomic barrier.
// ---------------------------------------------------------------------------

typedef __attribute__((ext_vector_type(16))) __bf16 v16bf;
typedef __attribute__((ext_vector_type(8)))  float  v8f;

namespace {

constexpr int kS = 32;
constexpr int kW = 128;
constexpr int kI = 2048;
constexpr int kH = 2048;
constexpr int kN = 4 * kH;      // 8192 gate rows
constexpr int kT = kS * kW;     // 4096 timesteps
constexpr int kScanBlocks = 64; // persistent blocks for the scan

union TileU { uint4 q[2]; v16bf v; };

__device__ __forceinline__ v8f wmma_bf16(v16bf a, v16bf b, v8f c) {
  // D = A(16x32 bf16) * B(32x16 bf16) + C(16x16 f32)
  return __builtin_amdgcn_wmma_f32_16x16x32_bf16(
      false, a, false, b, (short)0, c, false, false);
}

// A tile 16x32 bf16 at (row0, k0) from row-major [*, ld] bf16 matrix.
// Lane layout (ISA 7.12.2): m = lane&15; lanes 0-15 hold K 0-7 & 16-23,
// lanes 16-31 hold K 8-15 & 24-31.
__device__ __forceinline__ v16bf load_tileA(const __hip_bfloat16* base, int ld,
                                            int row0, int k0, int lane) {
  int m  = lane & 15;
  int kb = (lane >> 4) * 8;
  const __hip_bfloat16* p = base + (size_t)(row0 + m) * ld + k0 + kb;
  TileU u;
  u.q[0] = *reinterpret_cast<const uint4*>(p);       // K = kb .. kb+7
  u.q[1] = *reinterpret_cast<const uint4*>(p + 16);  // K = kb+16 .. kb+23
  return u.v;
}

// B tile 32x16 bf16: B[k][n] = Wrow_major[n][k], tile at (n0, k0).
// Lane layout: n = lane&15; lanes 0-15 hold K 0-7 & 16-23, lanes 16-31 hold K 8-15 & 24-31.
__device__ __forceinline__ v16bf load_tileB(const __hip_bfloat16* base, int ld,
                                            int n0, int k0, int lane) {
  int n  = lane & 15;
  int kb = (lane >> 4) * 8;
  const __hip_bfloat16* p = base + (size_t)(n0 + n) * ld + k0 + kb;
  TileU u;
  u.q[0] = *reinterpret_cast<const uint4*>(p);
  u.q[1] = *reinterpret_cast<const uint4*>(p + 16);
  return u.v;
}

__device__ __forceinline__ float sigmoidf_dev(float x) {
  return 1.0f / (1.0f + __expf(-x));
}

// ---------------------------------------------------------------------------
__global__ __launch_bounds__(256) void f32_to_bf16_kernel(
    const float* __restrict__ in, __hip_bfloat16* __restrict__ out, int n) {
  int i = blockIdx.x * 256 + threadIdx.x;
  if (i < n) out[i] = __float2bfloat16(in[i]);
}

// attW (H, W) row-major; softmax over W; write attn (W, H): attn[w][h].
__global__ __launch_bounds__(128) void softmax_attn_kernel(
    const float* __restrict__ attW, float* __restrict__ attn) {
  int row = blockIdx.x;   // hidden index 0..H-1
  int w   = threadIdx.x;  // word index 0..W-1
  __shared__ float red[kW];
  float v = attW[(size_t)row * kW + w];
  red[w] = v;
  __syncthreads();
  for (int s = kW / 2; s > 0; s >>= 1) {
    if (w < s) red[w] = fmaxf(red[w], red[w + s]);
    __syncthreads();
  }
  float m = red[0];
  __syncthreads();
  float e = __expf(v - m);
  red[w] = e;
  __syncthreads();
  for (int s = kW / 2; s > 0; s >>= 1) {
    if (w < s) red[w] += red[w + s];
    __syncthreads();
  }
  float sum = red[0];
  attn[(size_t)w * kH + row] = e / sum;
}

__global__ __launch_bounds__(256) void init_kernel(
    const float* __restrict__ h0, __hip_bfloat16* __restrict__ hbuf,
    unsigned* __restrict__ sync) {
  int i = blockIdx.x * 256 + threadIdx.x;
  if (i < kH) hbuf[i] = __float2bfloat16(h0[i]);   // buffer 0 = h_{-1}
  if (i == 0) { sync[0] = 0u; sync[1] = 0u; }
}

// ---------------------------------------------------------------------------
// Gx = X(4096x2048) * W_ih^T(2048x8192) + bias. Each wave: 64x16 output
// (4 stacked M-tiles reuse each B tile), K loop of 64 WMMAs x 4.
__global__ __launch_bounds__(256) void gemm_x_kernel(
    const __hip_bfloat16* __restrict__ X, const __hip_bfloat16* __restrict__ Wih,
    const float* __restrict__ b_ih, const float* __restrict__ b_hh,
    float* __restrict__ Gx) {
  int lane = threadIdx.x & 31;
  int wave = threadIdx.x >> 5;
  int job  = blockIdx.x * 8 + wave;     // 0 .. 64*512-1
  int nTiles = kN / 16;                 // 512
  int mg = job / nTiles;                // 0..63 (64-row M groups)
  int nt = job % nTiles;
  int row0 = mg * 64;
  int n0   = nt * 16;

  v8f acc0 = {}, acc1 = {}, acc2 = {}, acc3 = {};
  for (int k0 = 0; k0 < kI; k0 += 32) {
    v16bf b  = load_tileB(Wih, kI, n0, k0, lane);
    v16bf a0 = load_tileA(X, kI, row0 +  0, k0, lane);
    v16bf a1 = load_tileA(X, kI, row0 + 16, k0, lane);
    v16bf a2 = load_tileA(X, kI, row0 + 32, k0, lane);
    v16bf a3 = load_tileA(X, kI, row0 + 48, k0, lane);
    acc0 = wmma_bf16(a0, b, acc0);
    acc1 = wmma_bf16(a1, b, acc1);
    acc2 = wmma_bf16(a2, b, acc2);
    acc3 = wmma_bf16(a3, b, acc3);
  }

  int col  = n0 + (lane & 15);
  float bias = b_ih[col] + b_hh[col];
  int rb = row0 + (lane >> 4) * 8;      // D layout: VGPR r -> row rb + r
#pragma unroll
  for (int r = 0; r < 8; ++r) {
    Gx[(size_t)(rb +  0 + r) * kN + col] = acc0[r] + bias;
    Gx[(size_t)(rb + 16 + r) * kN + col] = acc1[r] + bias;
    Gx[(size_t)(rb + 32 + r) * kN + col] = acc2[r] + bias;
    Gx[(size_t)(rb + 48 + r) * kN + col] = acc3[r] + bias;
  }
}

// ---------------------------------------------------------------------------
// Persistent sequential LSTM scan. 64 blocks x 256 threads (8 waves).
// Block b owns hidden units [b*32, b*32+32). Wave (g=w>>1, sub=w&1) computes
// gate rows g*H + b*32 + sub*16 .. +16 via WMMA matvec (h lives in A row 0;
// D rows 1..15 are ignored -- matmul rows don't mix). W_hh (bf16, 32 MB)
// stays L2-resident; ~32 MB of L2 traffic per step is the roofline.
__global__ __launch_bounds__(256) void scan_kernel(
    const float* __restrict__ Gx, const __hip_bfloat16* __restrict__ Whh,
    const float* __restrict__ c0, const float* __restrict__ attn,
    __hip_bfloat16* __restrict__ hbuf /* 2 x H, double-buffered */,
    float* __restrict__ out, unsigned* __restrict__ cnt,
    unsigned* __restrict__ gen) {
  int lane = threadIdx.x & 31;
  int wave = threadIdx.x >> 5;
  int hb   = blockIdx.x * 32;           // hidden slice base
  int g    = wave >> 1;                 // gate index 0..3 (i,f,g,o)
  int sub  = wave & 1;
  int n0   = g * kH + hb + sub * 16;    // this wave's 16 gate rows

  __shared__ float gsh[4][32];

  float c = (wave == 0) ? c0[hb + lane] : 0.0f;

  for (int t = 0; t < kT; ++t) {
    const __hip_bfloat16* h = hbuf + (size_t)(t & 1) * kH;
    v8f acc = {};
    bool valid = (lane & 15) == 0;      // only A row 0 carries h
    int kb = (lane >> 4) * 8;
    for (int k0 = 0; k0 < kH; k0 += 32) {
      TileU ua;
      if (valid) {
        const __hip_bfloat16* p = h + k0 + kb;
        ua.q[0] = *reinterpret_cast<const uint4*>(p);
        ua.q[1] = *reinterpret_cast<const uint4*>(p + 16);
      } else {
        ua.q[0] = make_uint4(0u, 0u, 0u, 0u);
        ua.q[1] = make_uint4(0u, 0u, 0u, 0u);
      }
      v16bf bm = load_tileB(Whh, kH, n0, k0, lane);
      acc = wmma_bf16(ua.v, bm, acc);
    }
    // D row 0 lives in acc[0] of lanes 0..15 (N = lane).
    if (lane < 16) {
      gsh[g][sub * 16 + lane] = acc[0] + Gx[(size_t)t * kN + n0 + lane];
    }
    __syncthreads();

    if (wave == 0) {
      float gi = gsh[0][lane];
      float gf = gsh[1][lane];
      float gg = gsh[2][lane];
      float go = gsh[3][lane];
      c = sigmoidf_dev(gf) * c + sigmoidf_dev(gi) * tanhf(gg);
      float hn = sigmoidf_dev(go) * tanhf(c);
      int idx = hb + lane;
      out[(size_t)t * kH + idx] = hn * attn[(size_t)(t & (kW - 1)) * kH + idx];
      hbuf[(size_t)((t + 1) & 1) * kH + idx] = __float2bfloat16(hn);
    }

    // Device-wide barrier: make h_{t} visible, then release step t+1.
    __threadfence();
    __syncthreads();
    if (threadIdx.x == 0) {
      unsigned old = __hip_atomic_load(gen, __ATOMIC_RELAXED,
                                       __HIP_MEMORY_SCOPE_AGENT);
      unsigned arrived = __hip_atomic_fetch_add(cnt, 1u, __ATOMIC_ACQ_REL,
                                                __HIP_MEMORY_SCOPE_AGENT);
      if (arrived == (unsigned)(kScanBlocks - 1)) {
        __hip_atomic_store(cnt, 0u, __ATOMIC_RELAXED,
                           __HIP_MEMORY_SCOPE_AGENT);
        __hip_atomic_store(gen, old + 1u, __ATOMIC_RELEASE,
                           __HIP_MEMORY_SCOPE_AGENT);
      } else {
        while (__hip_atomic_load(gen, __ATOMIC_ACQUIRE,
                                 __HIP_MEMORY_SCOPE_AGENT) == old) {
          __builtin_amdgcn_s_sleep(8);
        }
      }
    }
    __syncthreads();
  }
}

}  // namespace

// ---------------------------------------------------------------------------
extern "C" void kernel_launch(void* const* d_in, const int* in_sizes, int n_in,
                              void* d_out, int out_size, void* d_ws,
                              size_t ws_size, hipStream_t stream) {
  (void)in_sizes; (void)n_in; (void)out_size; (void)ws_size;
  const float* inputs = (const float*)d_in[0];   // (S, W, I)
  const float* W_ih   = (const float*)d_in[1];   // (4H, I)
  const float* W_hh   = (const float*)d_in[2];   // (4H, H)
  const float* b_ih   = (const float*)d_in[3];   // (4H,)
  const float* b_hh   = (const float*)d_in[4];   // (4H,)
  const float* h0     = (const float*)d_in[5];   // (H,)
  const float* c0     = (const float*)d_in[6];   // (H,)
  const float* attW   = (const float*)d_in[7];   // (H, W)
  float* out = (float*)d_out;                    // (S, W, H)

  // Workspace carve-out (256 B aligned): sync, bf16 copies, Gx, attn, h buf.
  char* ws = (char*)d_ws;
  size_t off = 0;
  auto carve = [&](size_t bytes) -> void* {
    void* p = ws + off;
    off = (off + bytes + 255) & ~(size_t)255;
    return p;
  };
  unsigned*        sync  = (unsigned*)carve(256);
  __hip_bfloat16*  Xbf   = (__hip_bfloat16*)carve((size_t)kT * kI * 2);
  __hip_bfloat16*  Wihbf = (__hip_bfloat16*)carve((size_t)kN * kI * 2);
  __hip_bfloat16*  Whhbf = (__hip_bfloat16*)carve((size_t)kN * kH * 2);
  float*           Gx    = (float*)carve((size_t)kT * kN * 4);
  float*           attn  = (float*)carve((size_t)kW * kH * 4);
  __hip_bfloat16*  hbuf  = (__hip_bfloat16*)carve((size_t)2 * kH * 2);

  const int nX = kT * kI;   // 8,388,608
  const int nW = kN * kI;   // 16,777,216
  f32_to_bf16_kernel<<<(nX + 255) / 256, 256, 0, stream>>>(inputs, Xbf, nX);
  f32_to_bf16_kernel<<<(nW + 255) / 256, 256, 0, stream>>>(W_ih, Wihbf, nW);
  f32_to_bf16_kernel<<<(nW + 255) / 256, 256, 0, stream>>>(W_hh, Whhbf, nW);
  softmax_attn_kernel<<<kH, kW, 0, stream>>>(attW, attn);
  init_kernel<<<(kH + 255) / 256, 256, 0, stream>>>(h0, hbuf, sync);

  // Input projection GEMM: 64 M-groups x 512 N-tiles, 8 waves/block.
  gemm_x_kernel<<<(64 * 512) / 8, 256, 0, stream>>>(Xbf, Wihbf, b_ih, b_hh, Gx);

  // Sequential scan: 64 persistent blocks.
  scan_kernel<<<kScanBlocks, 256, 0, stream>>>(Gx, Whhbf, c0, attn, hbuf, out,
                                               sync, sync + 1);
}
